// SelfAttention2D_45200235823476
// MI455X (gfx1250) — matrix operand order repaired and checked by reading, not compiled, not run on hardware
//
#include <hip/hip_runtime.h>
#include <hip/hip_bf16.h>

typedef unsigned short u16;
typedef unsigned int   u32;

typedef __attribute__((ext_vector_type(16))) __bf16 v16bf;
typedef __attribute__((ext_vector_type(8)))  float  v8f;

#if __has_builtin(__builtin_amdgcn_global_load_async_to_lds_b128)
#define SA2D_ASYNC 1
typedef int v4i __attribute__((vector_size(16)));
typedef __attribute__((address_space(1))) v4i* as1_v4ip;
typedef __attribute__((address_space(3))) v4i* as3_v4ip;
#else
#define SA2D_ASYNC 0
#endif

union Frag16 {
    u32   u[8];
    v16bf v;
};

__device__ __forceinline__ u16 f2bf(float f) {
    union { float f; u32 u; } c; c.f = f;
    u32 u = c.u;
    u32 r = (u + 0x7FFFu + ((u >> 16) & 1u)) >> 16;   // round-to-nearest-even
    return (u16)r;
}

__device__ __forceinline__ v8f wmma_bf16(const Frag16& a, const Frag16& b, v8f c) {
    return __builtin_amdgcn_wmma_f32_16x16x32_bf16(
        /*neg_a=*/false, a.v, /*neg_b=*/false, b.v,
        /*c_mod=*/(short)0, c, /*reuse_a=*/false, /*reuse_b=*/false);
}

// A-fragment (16x32, 16-bit): lane holds row = l&15; VGPR v holds K pair at:
__device__ __forceinline__ int kidxA(int v, int lh) { return ((v >> 2) << 4) + (lh << 3) + ((v & 3) << 1); }
// B-fragment (32x16, 16-bit): lane holds col = l&15; VGPR v holds K pair at same K order as A:
__device__ __forceinline__ int kidxB(int v, int lh) { return ((v >> 2) << 4) + (lh << 3) + ((v & 3) << 1); }

// ---------------------------------------------------------------------------
// float -> bf16 elementwise convert (weights)
// ---------------------------------------------------------------------------
__global__ void sa2d_cvt_bf16(const float* __restrict__ in, u16* __restrict__ out, int n) {
    for (int i = blockIdx.x * blockDim.x + threadIdx.x; i < n; i += gridDim.x * blockDim.x)
        out[i] = f2bf(in[i]);
}

// ---------------------------------------------------------------------------
// GroupNorm: x (8,512,1024) f32 -> xnT (8,1024,512) bf16 (token-major for GEMM)
// one block per (batch, group); 16 channels x 1024 spatial contiguous
// ---------------------------------------------------------------------------
__global__ __launch_bounds__(256) void sa2d_groupnorm(
    const float* __restrict__ x, const float* __restrict__ gamma,
    const float* __restrict__ beta, u16* __restrict__ xnT) {
    const int bid = blockIdx.x;
    const int b = bid >> 5;       // 8 batches
    const int g = bid & 31;       // 32 groups
    const int tid = threadIdx.x;

    const float* xb = x + ((size_t)b * 512 + g * 16) * 1024;  // 16384 contiguous floats
    float s = 0.f, s2 = 0.f;
    for (int i = tid; i < 16384; i += 256) {
        float v = xb[i];
        s += v; s2 += v * v;
    }
    __shared__ float r1[256], r2[256];
    r1[tid] = s; r2[tid] = s2;
    __syncthreads();
    for (int st = 128; st > 0; st >>= 1) {
        if (tid < st) { r1[tid] += r1[tid + st]; r2[tid] += r2[tid + st]; }
        __syncthreads();
    }
    const float mu   = r1[0] * (1.0f / 16384.0f);
    const float var  = r2[0] * (1.0f / 16384.0f) - mu * mu;
    const float rstd = rsqrtf(var + 1e-5f);

    const int chl = tid & 15;
    const int ch  = g * 16 + chl;
    const float gam = gamma[ch], bet = beta[ch];
    const float* xc = x + ((size_t)b * 512 + ch) * 1024;
    u16* ob = xnT + (size_t)b * 1024 * 512 + ch;
    for (int s0 = tid >> 4; s0 < 1024; s0 += 16) {
        float v = (xc[s0] - mu) * rstd * gam + bet;
        ob[(size_t)s0 * 512] = f2bf(v);
    }
}

// ---------------------------------------------------------------------------
// bf16 WMMA GEMM: OUT(M,N) = A(M,512) * W(N,512)^T + bias
//   A row-major bf16 (K contiguous), W row-major (o,c) bf16 (K contiguous)
//   mode 0: store bf16 row-major (M,N)                       [qkv]
//   mode 1: store f32 to (b, n, s) layout + residual from x  [proj + residual]
// block = 256 thr = 8 waves (4 Mx2 N); block tile 128x128; wave tile 32x64
// everything is L2-resident -> fragments loaded straight from global
// ---------------------------------------------------------------------------
__global__ __launch_bounds__(256) void sa2d_gemm_bf16(
    const u16* __restrict__ A, const u16* __restrict__ W,
    const float* __restrict__ bias, int N, int mode,
    u16* __restrict__ outb, float* __restrict__ outf,
    const float* __restrict__ xres) {
    const int tid = threadIdx.x;
    const int l  = tid & 31;
    const int lr = l & 15;
    const int lh = l >> 4;
    const int w  = tid >> 5;
    const int wm = w >> 1;     // 0..3
    const int wn = w & 1;      // 0..1
    const int mbase = blockIdx.y * 128 + wm * 32;
    const int nbase = blockIdx.x * 128 + wn * 64;

    v8f acc[2][4];
#pragma unroll
    for (int i = 0; i < 2; i++)
#pragma unroll
        for (int j = 0; j < 4; j++)
#pragma unroll
            for (int e = 0; e < 8; e++) acc[i][j][e] = 0.f;

    for (int kb = 0; kb < 512; kb += 32) {
        Frag16 af[2], bf[4];
#pragma unroll
        for (int ms = 0; ms < 2; ms++) {
            const u16* ap = A + (size_t)(mbase + ms * 16 + lr) * 512 + kb;
#pragma unroll
            for (int v = 0; v < 8; v++)
                af[ms].u[v] = *(const u32*)(ap + kidxA(v, lh));
        }
#pragma unroll
        for (int ns = 0; ns < 4; ns++) {
            const u16* wp = W + (size_t)(nbase + ns * 16 + lr) * 512 + kb;
#pragma unroll
            for (int v = 0; v < 8; v++)
                bf[ns].u[v] = *(const u32*)(wp + kidxB(v, lh));
        }
#pragma unroll
        for (int ms = 0; ms < 2; ms++)
#pragma unroll
            for (int ns = 0; ns < 4; ns++)
                acc[ms][ns] = wmma_bf16(af[ms], bf[ns], acc[ms][ns]);
    }

#pragma unroll
    for (int ms = 0; ms < 2; ms++)
#pragma unroll
        for (int ns = 0; ns < 4; ns++) {
            const int n = nbase + ns * 16 + lr;
            const float bv = bias[n];
#pragma unroll
            for (int j = 0; j < 8; j++) {
                const int m = mbase + ms * 16 + j + 8 * lh;
                const float val = acc[ms][ns][j] + bv;
                if (mode == 0) {
                    outb[(size_t)m * N + n] = f2bf(val);
                } else {
                    const int b = m >> 10, sp = m & 1023;
                    const size_t idx = ((size_t)b * 512 + n) * 1024 + sp;
                    outf[idx] = val + xres[idx];
                }
            }
        }
}

// ---------------------------------------------------------------------------
// Flash attention, bf16 WMMA, wave32.
//   qkv: (b*1024, 1536) bf16; head dim d=64, hw=1024, scale=1/8
//   grid: 512 blocks = b(8) x head(8) x qblock(8); 256 thr = 8 waves x 16 q-rows
//   K tile 32 wide: staged global->LDS with GLOBAL_LOAD_ASYNC_TO_LDS_B128
//   (ASYNCcnt) when available; V tile transposed into LDS; S = Q^T K via WMMA;
//   online softmax with 16-lane half-wave shuffle reductions; P(16x32) -> LDS
//   -> one bf16 A fragment; O += P * V^T via 4 WMMA. Next tile prefetched.
// ---------------------------------------------------------------------------
__global__ __launch_bounds__(256) void sa2d_attention(
    const u16* __restrict__ qkv, u16* __restrict__ attn_out) {
    const int bid  = blockIdx.x;
    const int qblk = bid & 7;
    const int head = (bid >> 3) & 7;
    const int b    = bid >> 6;
    const int tid  = threadIdx.x;
    const int w    = tid >> 5;
    const int l    = tid & 31;
    const int lr   = l & 15;
    const int lh   = l >> 4;
    const int qtile = qblk * 128 + w * 16;

    const u16* base = qkv + (size_t)b * 1024 * 1536;

    __shared__ __align__(16) u16 sK[32 * 64];       // [kpos][dd] row-major
    __shared__ __align__(16) u16 sV[64 * 32];       // [dd][kpos] transposed
    __shared__ __align__(16) u16 sP[8][16 * 32];    // per-wave P tile

    // Q^T fragments (16 q-rows x 64 d) — loaded once, reused all iterations
    Frag16 aq[2];
#pragma unroll
    for (int f = 0; f < 2; f++) {
        const u16* qp = base + (size_t)(qtile + lr) * 1536 + head * 64 + f * 32;
#pragma unroll
        for (int v = 0; v < 8; v++)
            aq[f].u[v] = *(const u32*)(qp + kidxA(v, lh));
    }

    float rm[8], rl[8];
    v8f o[4];
#pragma unroll
    for (int j = 0; j < 8; j++) { rm[j] = -3.0e38f; rl[j] = 0.f; }
#pragma unroll
    for (int n = 0; n < 4; n++)
#pragma unroll
        for (int e = 0; e < 8; e++) o[n][e] = 0.f;

    const u16* kg = base + 512 + head * 64;
    const u16* vg = base + 1024 + head * 64;

    for (int kt = 0; kt < 1024; kt += 32) {
        __syncthreads();   // previous iteration's LDS reads done

        // ---- stage K tile: async DMA global -> LDS (one b128 per thread) ----
#if SA2D_ASYNC
        {
            const int krow = tid >> 3;            // 0..31
            const int kcp  = tid & 7;             // 8 x 16B per 128B row
            const u16* gsrc = kg + (size_t)(kt + krow) * 1536 + kcp * 8;
            __builtin_amdgcn_global_load_async_to_lds_b128(
                (as1_v4ip)(const void*)gsrc,
                (as3_v4ip)(void*)&sK[krow * 64 + kcp * 8], 0, 0);
        }
#endif
        // ---- stage V tile transposed (and K synchronously if no async) ----
#pragma unroll
        for (int i = 0; i < 4; i++) {
            const int uidx = tid + i * 256;       // 0..1023 dwords
            const int row = uidx >> 5;            // 0..31 kpos
            const int cp  = uidx & 31;            // col pair
            const int col = cp << 1;
#if !SA2D_ASYNC
            const u32 kk2 = *(const u32*)(kg + (size_t)(kt + row) * 1536 + col);
            ((u32*)sK)[row * 32 + cp] = kk2;
#endif
            const u32 vv = *(const u32*)(vg + (size_t)(kt + row) * 1536 + col);
            sV[col * 32 + row]       = (u16)(vv & 0xFFFFu);
            sV[(col + 1) * 32 + row] = (u16)(vv >> 16);
        }
        // ---- prefetch next K/V stripe into WGP cache ----
        if (kt + 32 < 1024) {
            const int prow = tid >> 3;
            const int pcol = (tid & 7) * 8;
            __builtin_prefetch(kg + (size_t)(kt + 32 + prow) * 1536 + pcol, 0, 1);
            __builtin_prefetch(vg + (size_t)(kt + 32 + prow) * 1536 + pcol, 0, 1);
        }
#if SA2D_ASYNC
#if __has_builtin(__builtin_amdgcn_s_wait_asynccnt)
        __builtin_amdgcn_s_wait_asynccnt(0);
#else
        asm volatile("s_wait_asynccnt 0x0" ::: "memory");
#endif
#endif
        __syncthreads();

        // ---- S(16x32) = Q^T * K, fp32 accum, 2 col-subtiles x 2 K-chunks ----
        v8f sc[2];
#pragma unroll
        for (int ns = 0; ns < 2; ns++) {
            Frag16 bk[2];
            const int kpl = ns * 16 + lr;
#pragma unroll
            for (int f = 0; f < 2; f++)
#pragma unroll
                for (int v = 0; v < 8; v++)
                    bk[f].u[v] = *(const u32*)(sK + kpl * 64 + f * 32 + kidxB(v, lh));
            v8f c;
#pragma unroll
            for (int e = 0; e < 8; e++) c[e] = 0.f;
            c = wmma_bf16(aq[0], bk[0], c);
            c = wmma_bf16(aq[1], bk[1], c);
            sc[ns] = c;
        }

        // ---- online softmax; rows j+8*lh live across 16-lane half-waves ----
        u16* pw = sP[w];
#pragma unroll
        for (int j = 0; j < 8; j++) {
            const float s0 = sc[0][j] * 0.125f;
            const float s1 = sc[1][j] * 0.125f;
            float t = fmaxf(s0, s1);
            t = fmaxf(t, __shfl_xor(t, 1, 32));
            t = fmaxf(t, __shfl_xor(t, 2, 32));
            t = fmaxf(t, __shfl_xor(t, 4, 32));
            t = fmaxf(t, __shfl_xor(t, 8, 32));
            const float mn = fmaxf(rm[j], t);
            const float alpha = __expf(rm[j] - mn);
            rm[j] = mn;
            const float p0 = __expf(s0 - mn);
            const float p1 = __expf(s1 - mn);
            float ps = p0 + p1;
            ps += __shfl_xor(ps, 1, 32);
            ps += __shfl_xor(ps, 2, 32);
            ps += __shfl_xor(ps, 4, 32);
            ps += __shfl_xor(ps, 8, 32);
            rl[j] = rl[j] * alpha + ps;
#pragma unroll
            for (int n = 0; n < 4; n++) o[n][j] *= alpha;
            const int row = j + 8 * lh;
            pw[row * 32 + lr]      = f2bf(p0);
            pw[row * 32 + 16 + lr] = f2bf(p1);
        }

        // ---- O(16x64) += P(16x32) * V^T(32x64) ----
        Frag16 ap;   // P tile is exactly one bf16 A fragment
#pragma unroll
        for (int v = 0; v < 8; v++)
            ap.u[v] = *(const u32*)(pw + lr * 32 + kidxA(v, lh));
#pragma unroll
        for (int n = 0; n < 4; n++) {
            Frag16 bv;
            const int dd = n * 16 + lr;
#pragma unroll
            for (int v = 0; v < 8; v++)
                bv.u[v] = *(const u32*)(sV + dd * 32 + kidxB(v, lh));
            o[n] = wmma_bf16(ap, bv, o[n]);
        }
    }

    // ---- epilogue: O /= l, store bf16 to (b*1024+q, 512) token-major ----
#pragma unroll
    for (int j = 0; j < 8; j++) {
        const float inv = 1.0f / rl[j];
        const int q = qtile + j + 8 * lh;
#pragma unroll
        for (int n = 0; n < 4; n++) {
            const int dd = head * 64 + n * 16 + lr;
            attn_out[(size_t)(b * 1024 + q) * 512 + dd] = f2bf(o[n][j] * inv);
        }
    }
}

// ---------------------------------------------------------------------------
extern "C" void kernel_launch(void* const* d_in, const int* in_sizes, int n_in,
                              void* d_out, int out_size, void* d_ws, size_t ws_size,
                              hipStream_t stream) {
    const float* x      = (const float*)d_in[0];   // (8,512,32,32)
    const float* gn_w   = (const float*)d_in[1];   // (512)
    const float* gn_b   = (const float*)d_in[2];   // (512)
    const float* qkv_w  = (const float*)d_in[3];   // (1536,512)
    const float* qkv_b  = (const float*)d_in[4];   // (1536)
    const float* proj_w = (const float*)d_in[5];   // (512,512)
    const float* proj_b = (const float*)d_in[6];   // (512)
    float* out = (float*)d_out;                    // (8,512,32,32)

    char* ws = (char*)d_ws;
    u16* xnT     = (u16*)(ws);                                // 8192x512   bf16 (8 MB)
    u16* wq_bf   = (u16*)(ws + 8388608);                      // 1536x512   bf16
    u16* wp_bf   = (u16*)(ws + 8388608 + 1572864);            // 512x512    bf16
    u16* qkv_buf = (u16*)(ws + 8388608 + 1572864 + 524288);   // 8192x1536  bf16 (24 MB)
    u16* att_buf = (u16*)(ws + 8388608 + 1572864 + 524288 + 25165824); // 8192x512 bf16

    // 1) weights -> bf16
    sa2d_cvt_bf16<<<1024, 256, 0, stream>>>(qkv_w, wq_bf, 1536 * 512);
    sa2d_cvt_bf16<<<1024, 256, 0, stream>>>(proj_w, wp_bf, 512 * 512);

    // 2) GroupNorm -> token-major bf16
    sa2d_groupnorm<<<256, 256, 0, stream>>>(x, gn_w, gn_b, xnT);

    // 3) QKV GEMM: (8192,512) x (1536,512)^T -> (8192,1536) bf16
    sa2d_gemm_bf16<<<dim3(12, 64), 256, 0, stream>>>(
        xnT, wq_bf, qkv_b, 1536, 0, qkv_buf, nullptr, nullptr);

    // 4) Flash attention (8 heads, d=64, 1024 tokens per batch)
    sa2d_attention<<<512, 256, 0, stream>>>(qkv_buf, att_buf);

    // 5) Proj GEMM + residual, fp32 channel-major store
    sa2d_gemm_bf16<<<dim3(4, 64), 256, 0, stream>>>(
        att_buf, wp_bf, proj_b, 512, 1, nullptr, out, x);
}
